// MultiHeadAttention_15393162789003
// MI455X (gfx1250) — 8-bit matrix operand order repaired and checked by reading, compiled, not run on hardware
//
#include <hip/hip_runtime.h>
#include <hip/hip_bf16.h>
#include <stdint.h>

typedef __attribute__((ext_vector_type(8))) int v8i;

#define EPS_F 1e-5f


__global__ __launch_bounds__(256)
void k_wabs_partial(const float* __restrict__ W, float* __restrict__ partial) {
  __shared__ float red[256];
  const size_t base = (size_t)blockIdx.x * 1024;
  float s = 0.f;
  for (int i = threadIdx.x; i < 1024; i += 256) s += fabsf(W[base + i]);
  red[threadIdx.x] = s;
  __syncthreads();
  for (int off = 128; off > 0; off >>= 1) {
    if ((int)threadIdx.x < off) red[threadIdx.x] += red[threadIdx.x + off];
    __syncthreads();
  }
  if (threadIdx.x == 0) partial[blockIdx.x] = red[0];
}

__global__ __launch_bounds__(256)
void k_wscale(const float* __restrict__ partial, float* __restrict__ wsc,
              float inv_count) {
  __shared__ float red[256];
  float s = 0.f;
  for (int i = threadIdx.x; i < 1024; i += 256) s += partial[i];
  red[threadIdx.x] = s;
  __syncthreads();
  for (int off = 128; off > 0; off >>= 1) {
    if ((int)threadIdx.x < off) red[threadIdx.x] += red[threadIdx.x + off];
    __syncthreads();
  }
  if (threadIdx.x == 0) {
    float mean = red[0] * inv_count;
    float dq = fmaxf(mean, EPS_F);
    wsc[0] = dq;
    wsc[1] = 1.0f / dq;
  }
}

__global__ __launch_bounds__(256)
void k_wquant(const float* __restrict__ W, const float* __restrict__ wsc,
              int8_t* __restrict__ Wq, int n) {
  int i = blockIdx.x * 256 + threadIdx.x;
  if (i >= n) return;
  float t = rintf(W[i] * wsc[1]);
  t = fminf(fmaxf(t, -1.f), 1.f);
  Wq[i] = (int8_t)t;
}


__global__ __launch_bounds__(256)
void k_act_quant(const float* __restrict__ X, int8_t* __restrict__ Xq,
                 float* __restrict__ dsx, int D) {
  const int row = blockIdx.x;
  const float* xp = X + (size_t)row * D;
  __shared__ float rs[256], rm[256];
  __shared__ float s_mul;
  float ss = 0.f, mx = 0.f;
  for (int i = threadIdx.x; i < D; i += 256) {
    float v = xp[i];
    ss += v * v;
    mx = fmaxf(mx, fabsf(v));
  }
  rs[threadIdx.x] = ss;
  rm[threadIdx.x] = mx;
  __syncthreads();
  for (int off = 128; off > 0; off >>= 1) {
    if ((int)threadIdx.x < off) {
      rs[threadIdx.x] += rs[threadIdx.x + off];
      rm[threadIdx.x] = fmaxf(rm[threadIdx.x], rm[threadIdx.x + off]);
    }
    __syncthreads();
  }
  if (threadIdx.x == 0) {
    float rms    = rsqrtf(rs[0] / (float)D + 1e-6f);
    float rowmax = fmaxf(rm[0] * rms, EPS_F);
    dsx[row] = rowmax * (1.0f / 127.0f);
    s_mul    = rms * (127.0f / rowmax);
  }
  __syncthreads();
  const float sc = s_mul;
  int8_t* op = Xq + (size_t)row * D;
  for (int i = threadIdx.x; i < D; i += 256) {
    float qv = rintf(xp[i] * sc);
    qv = fminf(fmaxf(qv, -128.f), 127.f);
    op[i] = (int8_t)qv;
  }
}


__device__ __forceinline__ v8i load_fragA(const int8_t* p) {
  v8i r; int4 t;
  t = *(const int4*)(p +  0); r[0] = t.x; r[1] = t.y; r[2] = t.z; r[3] = t.w;
  t = *(const int4*)(p + 32); r[4] = t.x; r[5] = t.y; r[6] = t.z; r[7] = t.w;
  return r;
}

__device__ __forceinline__ v8i load_fragB(const int8_t* p) {
  v8i r; int4 t;
  t = *(const int4*)(p +  0); r[0] = t.x; r[1] = t.y; r[2] = t.z; r[3] = t.w;
  t = *(const int4*)(p + 32); r[4] = t.x; r[5] = t.y; r[6] = t.z; r[7] = t.w;
  return r;
}

#define WMMA_IU8(A_, B_, C_) \
  __builtin_amdgcn_wmma_i32_16x16x64_iu8(true, (A_), true, (B_), (C_), false, false)

__global__ __launch_bounds__(256)
void k_bitgemm_iu8(const int8_t* __restrict__ A, const int8_t* __restrict__ W,
                   const float* __restrict__ sx, const float* __restrict__ swsc,
                   float* __restrict__ C, int M, int N, int K) {
  const int lane = threadIdx.x & 31;
  const int wave = threadIdx.x >> 5;
  const int m0 = blockIdx.x * 256 + wave * 32;
  const int n0 = blockIdx.y * 64;
  if (m0 >= M) return;
  const int lo   = lane & 15;
  const int kg8  = (lane >> 4) * 8;
  const int kg16 = (lane >> 4) * 16;
  const float sw = swsc[0];

  const int8_t* __restrict__ ar0 = A + (size_t)(m0 + lo) * K + kg16;
  const int8_t* __restrict__ ar1 = ar0 + (size_t)16 * K;
  const int8_t* __restrict__ w0 = W + (size_t)(n0 + lo) * K + kg16;
  const int8_t* __restrict__ w1 = w0 + (size_t)16 * K;
  const int8_t* __restrict__ w2 = w0 + (size_t)32 * K;
  const int8_t* __restrict__ w3 = w0 + (size_t)48 * K;

  v8i acc00 = {}; v8i acc01 = {}; v8i acc02 = {}; v8i acc03 = {};
  v8i acc10 = {}; v8i acc11 = {}; v8i acc12 = {}; v8i acc13 = {};

  for (int k0 = 0; k0 < K; k0 += 64) {
    v8i a0 = load_fragA(ar0 + k0);
    v8i a1 = load_fragA(ar1 + k0);
    v8i b0 = load_fragB(w0 + k0);
    v8i b1 = load_fragB(w1 + k0);
    v8i b2 = load_fragB(w2 + k0);
    v8i b3 = load_fragB(w3 + k0);
    if (k0 + 256 < K) {
      __builtin_prefetch(ar0 + k0 + 256, 0, 3);
      __builtin_prefetch(ar1 + k0 + 256, 0, 3);
    }
    acc00 = WMMA_IU8(a0, b0, acc00);
    acc10 = WMMA_IU8(a1, b0, acc10);
    acc01 = WMMA_IU8(a0, b1, acc01);
    acc11 = WMMA_IU8(a1, b1, acc11);
    acc02 = WMMA_IU8(a0, b2, acc02);
    acc12 = WMMA_IU8(a1, b2, acc12);
    acc03 = WMMA_IU8(a0, b3, acc03);
    acc13 = WMMA_IU8(a1, b3, acc13);
  }

  const int nCol = n0 + lo;
  {
    const int mBase = m0 + (lane >> 4) * 8;
    float scales[8];
#pragma unroll
    for (int r = 0; r < 8; ++r) scales[r] = sx[mBase + r] * sw;
#pragma unroll
    for (int r = 0; r < 8; ++r) {
      float* crow = C + (size_t)(mBase + r) * N + nCol;
      crow[0]  = (float)acc00[r] * scales[r];
      crow[16] = (float)acc01[r] * scales[r];
      crow[32] = (float)acc02[r] * scales[r];
      crow[48] = (float)acc03[r] * scales[r];
    }
  }
  {
    const int mBase = m0 + 16 + (lane >> 4) * 8;
    float scales[8];
#pragma unroll
    for (int r = 0; r < 8; ++r) scales[r] = sx[mBase + r] * sw;
#pragma unroll
    for (int r = 0; r < 8; ++r) {
      float* crow = C + (size_t)(mBase + r) * N + nCol;
      crow[0]  = (float)acc10[r] * scales[r];
      crow[16] = (float)acc11[r] * scales[r];
      crow[32] = (float)acc12[r] * scales[r];
      crow[48] = (float)acc13[r] * scales[r];
    }
  }
}


__global__ __launch_bounds__(128)
void k_attn(const float* __restrict__ Q, const float* __restrict__ Km,
            const float* __restrict__ V, float* __restrict__ ctx, int S) {
  const int token = blockIdx.x;
  const int b = token / S;
  const int s = token - b * S;
  __shared__ float Qs[1024], Ks[1024], Vs[1024], As[256];
  const size_t roff = (size_t)token * 1024;
  for (int i = threadIdx.x; i < 1024; i += 128) {
    Qs[i] = Q[roff + i];
    Ks[i] = Km[roff + i];
    Vs[i] = V[roff + i];
  }
  __syncthreads();
  for (int idx = threadIdx.x; idx < 256; idx += 128) {
    const int h = idx >> 4, t = idx & 15;
    float acc = 0.f;
#pragma unroll 8
    for (int d = 0; d < 64; ++d) acc += Qs[h * 64 + d] * Ks[t * 64 + d];
    As[idx] = acc * 0.125f;
  }
  __syncthreads();
  if ((int)threadIdx.x < 16) {
    const int h = threadIdx.x;
    float mx = As[h * 16];
    for (int t = 1; t < 16; ++t) mx = fmaxf(mx, As[h * 16 + t]);
    float sum = 0.f;
    for (int t = 0; t < 16; ++t) {
      float e = expf(As[h * 16 + t] - mx);
      As[h * 16 + t] = e;
      sum += e;
    }
    const float inv = 1.0f / sum;
    for (int t = 0; t < 16; ++t) As[h * 16 + t] *= inv;
  }
  __syncthreads();
  for (int idx = threadIdx.x; idx < 1024; idx += 128) {
    const int h = idx >> 6, d = idx & 63;
    float acc = 0.f;
#pragma unroll
    for (int t = 0; t < 16; ++t) acc += As[h * 16 + t] * Vs[t * 64 + d];
    ctx[(size_t)b * S * 1024 + (size_t)h * S * 64 + (size_t)s * 64 + d] = acc;
  }
}


extern "C" void kernel_launch(void* const* d_in, const int* in_sizes, int n_in,
                              void* d_out, int out_size, void* d_ws, size_t ws_size,
                              hipStream_t stream) {
  (void)in_sizes; (void)n_in; (void)out_size; (void)ws_size;
  const float* x = (const float*)d_in[0];
  const float* w[4] = {(const float*)d_in[1], (const float*)d_in[2],
                       (const float*)d_in[3], (const float*)d_in[4]};
  float* out = (float*)d_out;

  const int S = 4096, D = 1024;
  const int M = 2 * S;
  const size_t KK = (size_t)D * D;
  const size_t MK = (size_t)M * D;

  char* ws = (char*)d_ws;
  size_t o = 0;
  auto carve = [&](size_t bytes) -> void* {
    void* p = ws + o;
    o = (o + bytes + 255) & ~(size_t)255;
    return p;
  };
  int8_t* wq8[4];
  for (int i = 0; i < 4; ++i) wq8[i] = (int8_t*)carve(KK);
  float* part = (float*)carve(1024 * sizeof(float));
  float* wsc[4];
  for (int i = 0; i < 4; ++i) wsc[i] = (float*)carve(2 * sizeof(float));
  int8_t* x8  = (int8_t*)carve(MK);
  float*  sxv = (float*)carve((size_t)M * sizeof(float));
  int8_t* c8  = (int8_t*)carve(MK);
  float*  scv = (float*)carve((size_t)M * sizeof(float));
  float* qb = (float*)carve(MK * sizeof(float));
  float* kb = (float*)carve(MK * sizeof(float));
  float* vb = (float*)carve(MK * sizeof(float));
  float* cb = (float*)carve(MK * sizeof(float));

  for (int i = 0; i < 4; ++i) {
    k_wabs_partial<<<1024, 256, 0, stream>>>(w[i], part);
    k_wscale<<<1, 256, 0, stream>>>(part, wsc[i], 1.0f / (float)KK);
    k_wquant<<<(int)(KK / 256), 256, 0, stream>>>(w[i], wsc[i], wq8[i], (int)KK);
  }

  k_act_quant<<<M, 256, 0, stream>>>(x, x8, sxv, D);

  dim3 gg(M / 256, D / 64);
  k_bitgemm_iu8<<<gg, 256, 0, stream>>>(x8, wq8[0], sxv, wsc[0], qb, M, D, D);
  k_bitgemm_iu8<<<gg, 256, 0, stream>>>(x8, wq8[1], sxv, wsc[1], kb, M, D, D);
  k_bitgemm_iu8<<<gg, 256, 0, stream>>>(x8, wq8[2], sxv, wsc[2], vb, M, D, D);

  k_attn<<<M, 128, 0, stream>>>(qb, kb, vb, cb, S);

  k_act_quant<<<M, 256, 0, stream>>>(cb, c8, scv, D);
  k_bitgemm_iu8<<<gg, 256, 0, stream>>>(c8, wq8[3], scv, wsc[3], out, M, D, D);
}
